// MoELayer_47906065220076
// MI455X (gfx1250) — compile-verified, matrix operand order checked by reading, not run on hardware
//
#include <hip/hip_runtime.h>

// ---------------------------------------------------------------------------
// MoE layer for MI455X (gfx1250, wave32, WMMA, async global->LDS).
// Sparse top-2 expert execution with bf16 WMMA (f32 accumulate).
// Weights are pre-converted once per launch into workspace as bf16 in a
// WMMA-B *fragment-contiguous* layout: per (32-row K-block, column) the 16
// K-pair dwords are stored [half][p], so each lane fetches its whole B
// fragment with two global_load_b128's. x is pre-converted to bf16 and
// staged to LDS with async global->LDS B128 DMA.
// ---------------------------------------------------------------------------

#define D_MODEL 1024
#define H_DIM   4096
#define N_EXP   8
#define N_TOK   8192          // B*S
#define AUX_WF  0.01f

#define XS 1048               // LDS x-tile row stride (ushorts): 2096B rows, 16B aligned,
                              // 524 dwords -> rows 0..15 hit distinct banks (gcd(524,64)=4)
#define HS 136                // LDS h-tile row stride (ushorts): 272B rows, 16B aligned,
                              // 68 dwords -> gcd(68,64)=4 -> rows 0..15 conflict-free
#define LPAD 513              // convert-kernel LDS tile stride (dwords, odd)

typedef __attribute__((ext_vector_type(16))) __bf16 v16bf;
typedef __attribute__((ext_vector_type(8)))  float  v8f;

union ABTile { v16bf v; unsigned int u[8]; uint4 q[2]; };

__device__ __forceinline__ unsigned short f2bf(float f) {
    unsigned int u = __float_as_uint(f);
    u += 0x7FFFu + ((u >> 16) & 1u);           // round-to-nearest-even
    return (unsigned short)(u >> 16);
}
__device__ __forceinline__ unsigned int pack2bf(float lo, float hi) {
    return (unsigned int)f2bf(lo) | ((unsigned int)f2bf(hi) << 16);
}
// Branchless tanh-approx gelu: tanh(u/2) = (e^u - 1)/(e^u + 1), u clamped.
__device__ __forceinline__ float gelu_tanh(float v) {
    float u = 1.5957691216f * (v + 0.044715f * v * v * v);   // 2*sqrt(2/pi)*(...)
    u = fminf(fmaxf(u, -30.f), 30.f);
    const float ex = __expf(u);
    const float t  = (ex - 1.0f) * __frcp_rn(ex + 1.0f);
    return 0.5f * v * (1.0f + t);
}

#if defined(__has_builtin)
#if __has_builtin(__builtin_amdgcn_s_wait_asynccnt)
#define WAIT_ASYNC0() __builtin_amdgcn_s_wait_asynccnt(0)
#endif
#endif
#ifndef WAIT_ASYNC0
#define WAIT_ASYNC0() asm volatile("s_wait_asynccnt 0x0" ::: "memory")
#endif

// ---------------------------------------------------------------------------
__global__ void zero_kernel(float* __restrict__ p, int n) {
    int i = blockIdx.x * blockDim.x + threadIdx.x;
    if (i < n) p[i] = 0.f;
}

// Pre-convert weights f32 -> bf16 into the WMMA-B fragment-contiguous layout.
// Source: (E*Kdim) x COLS row-major f32.  One block handles a 32-row x 512-col
// tile: coalesced pair loads -> LDS -> coalesced fragment-ordered dword stores.
// dst dword index = (kblock*COLS + col)*16 + half*8 + p, holding K-pair
// rows (2*kpair, 2*kpair+1) where kpair = (p&3) + 4*half + 8*(p>>2).
__global__ __launch_bounds__(256) void convert_w_kernel(
    const float* __restrict__ src, unsigned int* __restrict__ dst, int cols)
{
    __shared__ unsigned int ld[16 * LPAD];
    const int b  = blockIdx.x;            // 32-row K-block index (global over E)
    const int cb = blockIdx.y * 512;      // column block base
    const int tid = threadIdx.x;

    for (int i = tid; i < 16 * 512; i += 256) {
        const int r2 = i >> 9, c = i & 511;
        const size_t s0 = ((size_t)b * 32 + 2 * r2) * cols + cb + c;
        ld[r2 * LPAD + c] = pack2bf(src[s0], src[s0 + cols]);
    }
    __syncthreads();
    unsigned int* dblk = dst + ((size_t)b * cols + cb) * 16;
    for (int j = tid; j < 16 * 512; j += 256) {
        const int col = j >> 4, q = j & 15;
        const int half = q >> 3, p = q & 7;
        const int kpair = (p & 3) + 4 * half + 8 * (p >> 2);
        dblk[(size_t)col * 16 + q] = ld[kpair * LPAD + col];
    }
}

// Pre-convert x f32 -> bf16 row-major (pairs contiguous along d).
__global__ __launch_bounds__(256) void convert_x_kernel(
    const float* __restrict__ src, unsigned int* __restrict__ dst, int n)
{
    int i = blockIdx.x * blockDim.x + threadIdx.x;
    if (i >= n) return;
    const float2 f = ((const float2*)src)[i];
    dst[i] = pack2bf(f.x, f.y);
}

// ---------------------------------------------------------------------------
// Router: one wave32 per token.
__global__ __launch_bounds__(256) void router_kernel(
    const float* __restrict__ x, const float* __restrict__ rw,
    const float* __restrict__ rb, float* __restrict__ probs,
    int* __restrict__ cnt, float* __restrict__ probsum,
    int* __restrict__ tok_list, float* __restrict__ gate_list)
{
    const int lane = threadIdx.x & 31;
    const int tok  = blockIdx.x * 8 + (threadIdx.x >> 5);
    if (tok >= N_TOK) return;

    float acc[N_EXP];
#pragma unroll
    for (int e = 0; e < N_EXP; ++e) acc[e] = 0.f;

    const float* xr = x + (size_t)tok * D_MODEL;
    for (int d = lane; d < D_MODEL; d += 32) {
        const float xv = xr[d];
        const float* r = rw + d * N_EXP;
#pragma unroll
        for (int e = 0; e < N_EXP; ++e) acc[e] += xv * r[e];
    }
#pragma unroll
    for (int e = 0; e < N_EXP; ++e) {
        float v = acc[e];
#pragma unroll
        for (int off = 16; off >= 1; off >>= 1) v += __shfl_xor(v, off, 32);
        acc[e] = v;
    }

    if (lane == 0) {
        float l[N_EXP], p[N_EXP];
        float mx = -1e30f;
#pragma unroll
        for (int e = 0; e < N_EXP; ++e) { l[e] = acc[e] + rb[e]; mx = fmaxf(mx, l[e]); }
        float s = 0.f;
#pragma unroll
        for (int e = 0; e < N_EXP; ++e) { p[e] = __expf(l[e] - mx); s += p[e]; }
        const float inv = 1.0f / s;
#pragma unroll
        for (int e = 0; e < N_EXP; ++e) {
            p[e] *= inv;
            probs[(size_t)tok * N_EXP + e] = p[e];
            atomicAdd(&probsum[e], p[e]);
        }
        int i0 = 0;
#pragma unroll
        for (int e = 1; e < N_EXP; ++e) if (l[e] > l[i0]) i0 = e;
        int i1 = (i0 == 0) ? 1 : 0;
#pragma unroll
        for (int e = 0; e < N_EXP; ++e) if (e != i0 && l[e] > l[i1]) i1 = e;
        const float e1 = __expf(l[i1] - l[i0]);
        const float g0 = 1.0f / (1.0f + e1);
        const float g1 = e1 * g0;

        int s0 = atomicAdd(&cnt[i0], 1);
        tok_list[i0 * N_TOK + s0] = tok;  gate_list[i0 * N_TOK + s0] = g0;
        int s1 = atomicAdd(&cnt[i1], 1);
        tok_list[i1 * N_TOK + s1] = tok;  gate_list[i1 * N_TOK + s1] = g1;
    }
}

// ---------------------------------------------------------------------------
__global__ void aux_kernel(const int* __restrict__ cnt,
                           const float* __restrict__ probsum,
                           float* __restrict__ aux_out)
{
    if (threadIdx.x == 0 && blockIdx.x == 0) {
        float s = 0.f;
#pragma unroll
        for (int e = 0; e < N_EXP; ++e)
            s += ((float)cnt[e] / (float)(N_TOK * 2)) * (probsum[e] / (float)N_TOK);
        *aux_out = (float)N_EXP * s * AUX_WF;
    }
}

// ---------------------------------------------------------------------------
// Expert FFN. One block = 16 gathered tokens of one expert, 8 waves.
// PRE=true : bf16 sources from workspace; B fragments fetched as 2x b128.
// PRE=false: on-the-fly f32->bf16 conversion (fallback when ws too small).
template <bool PRE>
__global__ __launch_bounds__(256) void ffn_kernel(
    const float* __restrict__ x,        const unsigned short* __restrict__ xbf,
    const float* __restrict__ w_up,     const uint4* __restrict__ wub,
    const float* __restrict__ b_up,
    const float* __restrict__ w_down,   const uint4* __restrict__ wdb,
    const float* __restrict__ b_down,
    const int*   __restrict__ cnt,      const int* __restrict__ tok_list,
    const float* __restrict__ gate_list, float* __restrict__ out)
{
    __shared__ __align__(16) unsigned short xs[16 * XS];  // 16 tokens x 1024 bf16
    __shared__ __align__(16) unsigned short hs[16 * HS];  // 16 tokens x 128  bf16

    const int e   = blockIdx.y;
    const int n_e = cnt[e];
    const int tb  = blockIdx.x * 16;
    if (tb >= n_e) return;

    const int tid  = threadIdx.x;
    const int lane = tid & 31;
    const int wave = tid >> 5;
    const int half = lane >> 4;
    const int l16  = lane & 15;

    const int* tl = tok_list + e * N_TOK;

    if (PRE) {
        // Async DMA: 16 rows x 2KB of bf16 x into LDS, 16B chunks.
        for (int i = tid; i < 16 * 128; i += 256) {
            const int m = i >> 7, c = i & 127;
            int slot = tb + m; if (slot >= n_e) slot = n_e - 1;
            const int tok = tl[slot];
            const unsigned dst  = (unsigned)(size_t)&xs[m * XS] + (unsigned)(c * 16);
            const unsigned goff = (unsigned)tok * (D_MODEL * 2) + (unsigned)(c * 16);
            asm volatile("global_load_async_to_lds_b128 %0, %1, %2 offset:0"
                         :: "v"(dst), "v"(goff), "s"(xbf) : "memory");
        }
        WAIT_ASYNC0();
    } else {
        for (int i = tid; i < 16 * D_MODEL; i += 256) {
            const int m = i >> 10, d = i & (D_MODEL - 1);
            int slot = tb + m; if (slot >= n_e) slot = n_e - 1;
            const int tok = tl[slot];
            xs[m * XS + d] = f2bf(x[(size_t)tok * D_MODEL + d]);
        }
    }
    __syncthreads();

    const v8f vzero = {0.f, 0.f, 0.f, 0.f, 0.f, 0.f, 0.f, 0.f};
    v8f acc[8];
#pragma unroll
    for (int t = 0; t < 8; ++t) acc[t] = vzero;

    const float* wu  = w_up   + (size_t)e * D_MODEL * H_DIM;
    const float* wd  = w_down + (size_t)e * H_DIM * D_MODEL;
    const int nbase = wave * 128;

    for (int hb = 0; hb < H_DIM; hb += 128) {
        // ---- up: this wave's 16x16 h tile at column hb + wave*16 + l16 ----
        const int hcol = hb + wave * 16 + l16;
        v8f hc = vzero;
        for (int kb = 0; kb < D_MODEL; kb += 32) {
            ABTile a, b;
#pragma unroll
            for (int p = 0; p < 8; ++p) {
                const int K = 2 * p + ((p >= 4) ? 8 : 0) + 8 * half;   // even
                a.u[p] = *(const unsigned int*)&xs[l16 * XS + kb + K];
            }
            if (PRE) {
                // fragment-contiguous: 2x b128 per B fragment
                const size_t kbg  = ((size_t)e * D_MODEL + kb) >> 5;
                const size_t idx4 = ((kbg * H_DIM + hcol) * 16 + half * 8) >> 2;
                b.q[0] = wub[idx4];
                b.q[1] = wub[idx4 + 1];
            } else {
#pragma unroll
                for (int p = 0; p < 8; ++p) {
                    const int K = 2 * p + ((p >= 4) ? 8 : 0) + 8 * half;
                    b.u[p] = pack2bf(wu[(size_t)(kb + K)     * H_DIM + hcol],
                                     wu[(size_t)(kb + K + 1) * H_DIM + hcol]);
                }
            }
            hc = __builtin_amdgcn_wmma_f32_16x16x32_bf16(false, a.v, false, b.v,
                                                         (short)0, hc, false, false);
        }
        const float bu = b_up[e * H_DIM + hcol];

        __syncthreads();                          // prior down-phase done with hs
#pragma unroll
        for (int r = 0; r < 8; ++r)
            hs[(r + 8 * half) * HS + wave * 16 + l16] = f2bf(gelu_tanh(hc[r] + bu));
        __syncthreads();                          // hs visible to all waves

        if (hb + 128 < H_DIM) {                   // warm next w_down panel
            if (PRE) {
                const size_t kbg = ((size_t)e * H_DIM + hb + 128) >> 5;
                __builtin_prefetch(&wdb[((kbg * D_MODEL + nbase + lane) * 16) >> 2], 0, 1);
            } else {
                __builtin_prefetch(&wd[(size_t)(hb + 128) * D_MODEL + nbase + lane * 4], 0, 1);
            }
        }

        // ---- down: out[16, nbase..nbase+128) += gelu(h) @ w_down panel ----
        for (int ks = 0; ks < 4; ++ks) {
            ABTile a;
#pragma unroll
            for (int p = 0; p < 8; ++p) {
                const int K = 2 * p + ((p >= 4) ? 8 : 0) + 8 * half;
                a.u[p] = *(const unsigned int*)&hs[l16 * HS + ks * 32 + K];
            }
            const int krow = hb + ks * 32;
            const size_t kbg = ((size_t)e * H_DIM + krow) >> 5;
#pragma unroll
            for (int nt = 0; nt < 8; ++nt) {
                ABTile b;
                const int col = nbase + nt * 16 + l16;
                if (PRE) {
                    const size_t idx4 = ((kbg * D_MODEL + col) * 16 + half * 8) >> 2;
                    b.q[0] = wdb[idx4];
                    b.q[1] = wdb[idx4 + 1];
                } else {
#pragma unroll
                    for (int p = 0; p < 8; ++p) {
                        const int K = 2 * p + ((p >= 4) ? 8 : 0) + 8 * half;
                        b.u[p] = pack2bf(wd[(size_t)(krow + K)     * D_MODEL + col],
                                         wd[(size_t)(krow + K + 1) * D_MODEL + col]);
                    }
                }
                acc[nt] = __builtin_amdgcn_wmma_f32_16x16x32_bf16(false, a.v, false, b.v,
                                                                  (short)0, acc[nt], false, false);
            }
        }
    }

    // ---- combine: (acc + b_down) * gate -> atomic add into output ----
    const float* gl = gate_list + e * N_TOK;
#pragma unroll
    for (int nt = 0; nt < 8; ++nt) {
        const int col = nbase + nt * 16 + l16;
        const float bd = b_down[e * D_MODEL + col];
#pragma unroll
        for (int r = 0; r < 8; ++r) {
            const int slot = tb + r + 8 * half;   // C-tile row M = r + 8*laneHalf
            if (slot < n_e) {
                const int tok = tl[slot];
                atomicAdd(&out[(size_t)tok * D_MODEL + col], (acc[nt][r] + bd) * gl[slot]);
            }
        }
    }
}

// ---------------------------------------------------------------------------
extern "C" void kernel_launch(void* const* d_in, const int* in_sizes, int n_in,
                              void* d_out, int out_size, void* d_ws, size_t ws_size,
                              hipStream_t stream)
{
    const float* x   = (const float*)d_in[0];
    const float* rw  = (const float*)d_in[1];
    const float* rb  = (const float*)d_in[2];
    const float* wup = (const float*)d_in[3];
    const float* bup = (const float*)d_in[4];
    const float* wdn = (const float*)d_in[5];
    const float* bdn = (const float*)d_in[6];

    float* out   = (float*)d_out;                        // (B,S,D)
    float* aux   = out + (size_t)N_TOK * D_MODEL;        // scalar
    float* probs = aux + 1;                              // (B,S,E)

    // Workspace layout (bytes):
    //   [0)      cnt       : 8 int
    //   [32)     probsum   : 8 float
    //   [64)     tok_list  : 8*8192 int   (256 KB)
    //   [+256K)  gate_list : 8*8192 float (256 KB)
    //   [524352) xbf       : 8192*1024 bf16                       (16 MB)
    //   [...]    wub       : fragment-contiguous bf16 w_up dwords (64 MB)
    //   [...]    wdb       : fragment-contiguous bf16 w_down      (64 MB)
    char* ws = (char*)d_ws;
    int*   cnt      = (int*)ws;
    float* probsum  = (float*)(ws + 32);
    int*   tok_list = (int*)(ws + 64);
    float* gate     = (float*)(ws + 64 + N_EXP * N_TOK * 4);
    const size_t routing_bytes = 64 + 2ull * N_EXP * N_TOK * 4;          // 524,352
    unsigned short* xbf = (unsigned short*)(ws + routing_bytes);
    const size_t xbf_bytes = (size_t)N_TOK * D_MODEL * 2;                // 16 MB
    unsigned int* wub = (unsigned int*)(ws + routing_bytes + xbf_bytes);
    const size_t w_dwords = (size_t)N_EXP * (D_MODEL / 2) * H_DIM;       // 16.78M
    unsigned int* wdb = wub + w_dwords;
    const size_t need = routing_bytes + xbf_bytes + 2 * w_dwords * 4;

    const int n_out = N_TOK * D_MODEL;
    zero_kernel<<<(n_out + 255) / 256, 256, 0, stream>>>(out, n_out);
    zero_kernel<<<1, 64, 0, stream>>>((float*)d_ws, 2 * N_EXP);

    router_kernel<<<N_TOK / 8, 256, 0, stream>>>(x, rw, rb, probs, cnt, probsum,
                                                 tok_list, gate);
    aux_kernel<<<1, 32, 0, stream>>>(cnt, probsum, aux);

    dim3 grid(N_TOK / 16, N_EXP);
    if (ws_size >= need) {
        convert_x_kernel<<<(N_TOK * D_MODEL / 2 + 255) / 256, 256, 0, stream>>>(
            x, (unsigned int*)xbf, N_TOK * D_MODEL / 2);
        // w_up: (E*D) x H -> 256 K-blocks x 8 col-blocks
        dim3 gu(N_EXP * D_MODEL / 32, H_DIM / 512);
        convert_w_kernel<<<gu, 256, 0, stream>>>(wup, wub, H_DIM);
        // w_down: (E*H) x D -> 1024 K-blocks x 2 col-blocks
        dim3 gd(N_EXP * H_DIM / 32, D_MODEL / 512);
        convert_w_kernel<<<gd, 256, 0, stream>>>(wdn, wdb, D_MODEL);
        ffn_kernel<true><<<grid, 256, 0, stream>>>(x, xbf, wup, (const uint4*)wub, bup,
                                                   wdn, (const uint4*)wdb, bdn, cnt,
                                                   tok_list, gate, out);
    } else {
        ffn_kernel<false><<<grid, 256, 0, stream>>>(x, xbf, wup, (const uint4*)wub, bup,
                                                    wdn, (const uint4*)wdb, bdn, cnt,
                                                    tok_list, gate, out);
    }
}
